// MultiHeadAttention_12111807774992
// MI455X (gfx1250) — matrix operand order repaired and checked by reading, not compiled, not run on hardware
//
#include <hip/hip_runtime.h>

#define EMBED 1024
#define HEADS 16
#define HDIM  64
#define LSEQ  2048
#define BATCH 4

typedef __attribute__((ext_vector_type(16))) _Float16 v16h;
typedef __attribute__((ext_vector_type(8)))  float    v8f;

// ---------------------------------------------------------------------------
// WMMA operand builders, derived from CDNA5 ISA 7.12.2 VGPR layouts (wave32).
// A (16x32, MxK, 16-bit): lane m=lane&15, half=lane>>4.
//   VGPR j holds K pair: k = (j<4 ? 0 : 16) + half*8 + (j&3)*2
// B (32x16, KxN, 16-bit): lane n=lane&15, half=lane>>4.
//   VGPR j holds K pair: k = (j<4 ? 0 : 16) + half*8 + (j&3)*2   (column n)
// C/D (16x16 f32): lane n=lane&15; VGPR r holds row m = r + 8*(lane>>4).
// ---------------------------------------------------------------------------

static __device__ __forceinline__ v8f wmma_f16(v16h a, v16h b, v8f c) {
  return __builtin_amdgcn_wmma_f32_16x16x32_f16(false, a, false, b,
                                                (short)0, c, false, false);
}

// A tile from row-major f16 source (leading dim ld, in elements)
static __device__ __forceinline__ v16h load_A(const _Float16* src, int ld) {
  const int lane = threadIdx.x & 31;
  const int m = lane & 15, half = lane >> 4;
  v16h a;
#pragma unroll
  for (int j = 0; j < 8; ++j) {
    const int k = ((j & 4) << 2) + half * 8 + ((j & 3) << 1);
    a[2 * j]     = src[m * ld + k];
    a[2 * j + 1] = src[m * ld + k + 1];
  }
  return a;
}

// A tile from row-major f32 source, scaled, converted to f16
static __device__ __forceinline__ v16h load_A_f32(const float* src, int ld,
                                                  float scale) {
  const int lane = threadIdx.x & 31;
  const int m = lane & 15, half = lane >> 4;
  v16h a;
#pragma unroll
  for (int j = 0; j < 8; ++j) {
    const int k = ((j & 4) << 2) + half * 8 + ((j & 3) << 1);
    a[2 * j]     = (_Float16)(src[m * ld + k] * scale);
    a[2 * j + 1] = (_Float16)(src[m * ld + k + 1] * scale);
  }
  return a;
}

// B tile (KxN) from an N-major source: srcT[n*ld + k] == B[k][n]
static __device__ __forceinline__ v16h load_B(const _Float16* srcT, int ld) {
  const int lane = threadIdx.x & 31;
  const int n = lane & 15, half = lane >> 4;
  v16h b;
#pragma unroll
  for (int j = 0; j < 8; ++j) {
    const int k = ((j & 4) << 2) + half * 8 + ((j & 3) << 1);
    b[2 * j]     = srcT[n * ld + k];
    b[2 * j + 1] = srcT[n * ld + k + 1];
  }
  return b;
}

static __device__ __forceinline__ v16h load_B_f32(const float* srcT, int ld) {
  const int lane = threadIdx.x & 31;
  const int n = lane & 15, half = lane >> 4;
  v16h b;
#pragma unroll
  for (int j = 0; j < 8; ++j) {
    const int k = ((j & 4) << 2) + half * 8 + ((j & 3) << 1);
    b[2 * j]     = (_Float16)srcT[n * ld + k];
    b[2 * j + 1] = (_Float16)srcT[n * ld + k + 1];
  }
  return b;
}

// ---------------------------------------------------------------------------
// Stage 1: per-head projection  y = x @ W.T  (W is 64x64, shared over heads).
// One wave handles one (b,l): 16 heads = 16 rows of a 16x64 GEMM (8 WMMAs).
// mode 0 -> out[B,H,L,D] (Q,K); mode 1 -> out[B,H,D,L] (V, transposed so the
// later P·V B-operand fetches are contiguous). Score scale 2^-5 folded into Q.
// ---------------------------------------------------------------------------
__global__ __launch_bounds__(256) void proj_kernel(
    const float* __restrict__ x, const float* __restrict__ W,
    _Float16* __restrict__ out, int transposed, float scale) {
  const int wave = threadIdx.x >> 5;
  const int bl = blockIdx.x * 8 + wave;  // 0 .. B*L-1
  const float* xr = x + (size_t)bl * EMBED;

  v16h a0 = load_A_f32(xr + 0, HDIM, scale);   // rows = heads, ld = 64
  v16h a1 = load_A_f32(xr + 32, HDIM, scale);

  v8f acc[4] = {};
#pragma unroll
  for (int nt = 0; nt < 4; ++nt) {
    acc[nt] = wmma_f16(a0, load_B_f32(W + nt * 16 * HDIM + 0, HDIM), acc[nt]);
    acc[nt] = wmma_f16(a1, load_B_f32(W + nt * 16 * HDIM + 32, HDIM), acc[nt]);
  }

  const int lane = threadIdx.x & 31;
  const int n = lane & 15, half = lane >> 4;
  const int b = bl >> 11, l = bl & (LSEQ - 1);
#pragma unroll
  for (int nt = 0; nt < 4; ++nt) {
#pragma unroll
    for (int r = 0; r < 8; ++r) {
      const int h = r + 8 * half;   // C-tile row == head
      const int d = nt * 16 + n;    // C-tile col == output dim
      const size_t idx =
          transposed ? (((size_t)(b * HEADS + h) * HDIM + d) * LSEQ + l)
                     : (((size_t)(b * HEADS + h) * LSEQ + l) * HDIM + d);
      out[idx] = (_Float16)acc[nt][r];
    }
  }
}

// ---------------------------------------------------------------------------
// Stage 2: flash attention. Block = 8 waves = 128 query rows of one (b,h).
// K/V staged cooperatively in LDS per 32-key block; each wave: 4 WMMAs for
// S = Q·K^T, online softmax (f32, shfl-xor row reductions), P through LDS
// (C-layout -> A-layout), 4 WMMAs for O += P·V.
// ---------------------------------------------------------------------------
__global__ __launch_bounds__(256) void attn_kernel(
    const _Float16* __restrict__ Qp, const _Float16* __restrict__ Kp,
    const _Float16* __restrict__ Vt, _Float16* __restrict__ Yp) {
  __shared__ alignas(16) _Float16 sK[32 * HDIM];  // [key][d]
  __shared__ alignas(16) _Float16 sV[HDIM * 32];  // [d][key]
  __shared__ alignas(16) _Float16 sP[8][16 * 32]; // per-wave P tile

  const int t = threadIdx.x;
  const int wave = t >> 5;
  const int lane = t & 31;
  const int n = lane & 15, half = lane >> 4;

  const int h = blockIdx.y, b = blockIdx.z;
  const int bh = b * HEADS + h;
  const int q0 = blockIdx.x * 128 + wave * 16;

  const _Float16* Kb = Kp + (size_t)bh * LSEQ * HDIM;
  const _Float16* Vb = Vt + (size_t)bh * HDIM * LSEQ;

  const _Float16* Qw = Qp + ((size_t)bh * LSEQ + q0) * HDIM;
  v16h aq0 = load_A(Qw + 0, HDIM);
  v16h aq1 = load_A(Qw + 32, HDIM);

  v8f o[4] = {};
  float mrow[8], lrow[8];
#pragma unroll
  for (int r = 0; r < 8; ++r) { mrow[r] = -1e30f; lrow[r] = 0.0f; }

  for (int kb = 0; kb < LSEQ; kb += 32) {
    // cooperative staging: 8 bytes (4 f16) per transfer, 2 per thread per buf
    for (int i = t; i < 512; i += 256) {
      ((uint2*)sK)[i] = ((const uint2*)(Kb + (size_t)kb * HDIM))[i];
      const int d = i >> 3, k4 = i & 7;
      ((uint2*)sV)[i] = *(const uint2*)(Vb + (size_t)d * LSEQ + kb + k4 * 4);
    }
    __syncthreads();

    // S = Q·K^T : two 16x16 tiles over this 32-key block
    v8f s[2];
#pragma unroll
    for (int nt = 0; nt < 2; ++nt) {
      v8f a = {};
      a = wmma_f16(aq0, load_B(sK + nt * 16 * HDIM + 0, HDIM), a);
      a = wmma_f16(aq1, load_B(sK + nt * 16 * HDIM + 32, HDIM), a);
      s[nt] = a;
    }

    // online softmax (scale already folded into Q)
    _Float16* pw = sP[wave];
#pragma unroll
    for (int r = 0; r < 8; ++r) {
      const float v0 = s[0][r], v1 = s[1][r];
      float vmax = fmaxf(v0, v1);
      vmax = fmaxf(vmax, __shfl_xor(vmax, 1, 32));
      vmax = fmaxf(vmax, __shfl_xor(vmax, 2, 32));
      vmax = fmaxf(vmax, __shfl_xor(vmax, 4, 32));
      vmax = fmaxf(vmax, __shfl_xor(vmax, 8, 32));
      const float mnew = fmaxf(mrow[r], vmax);
      const float p0 = __expf(v0 - mnew);
      const float p1 = __expf(v1 - mnew);
      float ps = p0 + p1;
      ps += __shfl_xor(ps, 1, 32);
      ps += __shfl_xor(ps, 2, 32);
      ps += __shfl_xor(ps, 4, 32);
      ps += __shfl_xor(ps, 8, 32);
      const float alpha = __expf(mrow[r] - mnew);
      lrow[r] = lrow[r] * alpha + ps;
      mrow[r] = mnew;
#pragma unroll
      for (int dt = 0; dt < 4; ++dt) o[dt][r] = o[dt][r] * alpha;
      const int row = r + 8 * half;
      pw[row * 32 + n]      = (_Float16)p0;
      pw[row * 32 + 16 + n] = (_Float16)p1;
    }
    // wave-local LDS RAW: DS ops are in-order per wave; stop compiler
    // reordering and drain DScnt before re-reading sP in A-layout.
    __builtin_amdgcn_wave_barrier();
    asm volatile("s_wait_dscnt 0x0" ::: "memory");

    v16h ap = load_A(pw, 32);
#pragma unroll
    for (int dt = 0; dt < 4; ++dt)
      o[dt] = wmma_f16(ap, load_B(sV + dt * 16 * 32, 32), o[dt]);
    __syncthreads();
  }

  float inv[8];
#pragma unroll
  for (int r = 0; r < 8; ++r) inv[r] = 1.0f / lrow[r];
#pragma unroll
  for (int dt = 0; dt < 4; ++dt) {
#pragma unroll
    for (int r = 0; r < 8; ++r) {
      const int q = q0 + r + 8 * half;
      const int d = dt * 16 + n;
      Yp[((size_t)b * LSEQ + q) * EMBED + h * HDIM + d] =
          (_Float16)(o[dt][r] * inv[r]);
    }
  }
}

// ---------------------------------------------------------------------------
// Stage 3: out = Y @ Wo.T + bo (f32 output). Block tile 128x64, Wo K-chunks
// staged f32->f16 in LDS; 4 WMMAs per chunk per wave, 32 chunks.
// ---------------------------------------------------------------------------
__global__ __launch_bounds__(256) void outproj_kernel(
    const _Float16* __restrict__ Y, const float* __restrict__ Wo,
    const float* __restrict__ bo, float* __restrict__ out) {
  __shared__ alignas(16) _Float16 sW[64 * 32];  // [o][e] chunk, N-major

  const int t = threadIdx.x;
  const int wave = t >> 5;
  const int lane = t & 31;
  const int n = lane & 15, half = lane >> 4;

  const int m0 = blockIdx.y * 128 + wave * 16;  // row tile of this wave
  const int o0 = blockIdx.x * 64;               // col tile of this block

  v8f acc[4] = {};
  for (int e0 = 0; e0 < EMBED; e0 += 32) {
    for (int i = t; i < 512; i += 256) {
      const int nn = i >> 3, k4 = i & 7;
      const float4 w =
          *(const float4*)(Wo + (size_t)(o0 + nn) * EMBED + e0 + k4 * 4);
      _Float16* dst = sW + nn * 32 + k4 * 4;
      dst[0] = (_Float16)w.x; dst[1] = (_Float16)w.y;
      dst[2] = (_Float16)w.z; dst[3] = (_Float16)w.w;
    }
    __syncthreads();
    v16h a = load_A(Y + (size_t)m0 * EMBED + e0, EMBED);
#pragma unroll
    for (int nt = 0; nt < 4; ++nt)
      acc[nt] = wmma_f16(a, load_B(sW + nt * 16 * 32, 32), acc[nt]);
    __syncthreads();
  }
#pragma unroll
  for (int nt = 0; nt < 4; ++nt) {
    const float bias = bo[o0 + nt * 16 + n];
#pragma unroll
    for (int r = 0; r < 8; ++r)
      out[(size_t)(m0 + r + 8 * half) * EMBED + o0 + nt * 16 + n] =
          acc[nt][r] + bias;
  }
}

// ---------------------------------------------------------------------------
extern "C" void kernel_launch(void* const* d_in, const int* in_sizes, int n_in,
                              void* d_out, int out_size, void* d_ws,
                              size_t ws_size, hipStream_t stream) {
  (void)in_sizes; (void)n_in; (void)out_size; (void)ws_size;
  const float* values  = (const float*)d_in[0];
  const float* keys    = (const float*)d_in[1];
  const float* queries = (const float*)d_in[2];
  const float* Wv = (const float*)d_in[3];
  const float* Wk = (const float*)d_in[4];
  const float* Wq = (const float*)d_in[5];
  const float* Wo = (const float*)d_in[6];
  const float* bo = (const float*)d_in[7];
  float* out = (float*)d_out;

  const size_t SEG = (size_t)BATCH * HEADS * LSEQ * HDIM * sizeof(_Float16); // 16MB
  char* ws = (char*)d_ws;
  _Float16* Qp = (_Float16*)(ws + 0 * SEG);
  _Float16* Kp = (_Float16*)(ws + 1 * SEG);
  _Float16* Vt = (_Float16*)(ws + 2 * SEG);  // [B,H,D,L]
  _Float16* Yp = (_Float16*)(ws + 3 * SEG);  // [B,L,E]

  const int projBlocks = BATCH * LSEQ / 8;  // 1024
  // score scale 1/sqrt(EMBED) = 2^-5 folded exactly into Q projection
  proj_kernel<<<projBlocks, 256, 0, stream>>>(queries, Wq, Qp, 0, 0.03125f);
  proj_kernel<<<projBlocks, 256, 0, stream>>>(keys,    Wk, Kp, 0, 1.0f);
  proj_kernel<<<projBlocks, 256, 0, stream>>>(values,  Wv, Vt, 1, 1.0f);

  attn_kernel<<<dim3(LSEQ / 128, HEADS, BATCH), 256, 0, stream>>>(Qp, Kp, Vt, Yp);

  outproj_kernel<<<dim3(EMBED / 64, BATCH * LSEQ / 128), 256, 0, stream>>>(
      Yp, Wo, bo, out);
}
